// AKT_36764920054295
// MI455X (gfx1250) — matrix operand order repaired and checked by reading, not compiled, not run on hardware
//
#include <hip/hip_runtime.h>
#include <hip/hip_bf16.h>

typedef unsigned short u16;
typedef __bf16 bf16_t;
typedef bf16_t v16bf __attribute__((ext_vector_type(16)));
typedef float  v8f   __attribute__((ext_vector_type(8)));

#define WMMA_BF16(a, b, c) \
  __builtin_amdgcn_wmma_f32_16x16x32_bf16(false, (a), false, (b), (short)0, (c), false, false)

union FragBF {
  v16bf v;
  uint4 q[2];
  u16   u[16];
};
union U16x8 {
  uint4 q;
  u16   u[8];
};
union U16x4 {
  uint2 q;
  u16   u[4];
};

__device__ __forceinline__ u16 f_to_bf(float f) {
  unsigned int x = __float_as_uint(f);
  x += 0x7FFFu + ((x >> 16) & 1u);   // round-to-nearest-even
  return (u16)(x >> 16);
}
__device__ __forceinline__ float bf_to_f(u16 u) {
  return __uint_as_float(((unsigned int)u) << 16);
}

// ---------------------------------------------------------------------------
// async global->LDS copy support (gfx1250): guarded by __has_builtin.
// arg0 = AS(1) ptr to int __attribute__((vector_size(16))), arg1 = AS(3) ptr,
// then imm offset, cpol.
// ---------------------------------------------------------------------------
#if defined(__has_builtin)
#if __has_builtin(__builtin_amdgcn_global_load_async_to_lds_b128)
#define HAVE_ASYNC_LDS 1
#endif
#endif
#ifndef HAVE_ASYNC_LDS
#define HAVE_ASYNC_LDS 0
#endif

#if HAVE_ASYNC_LDS
typedef int v4i_gcc __attribute__((vector_size(16)));
typedef __attribute__((address_space(1))) v4i_gcc* gptr128;
typedef __attribute__((address_space(3))) v4i_gcc* lptr128;
__device__ __forceinline__ void async_cp_b128(const u16* g, u16* l) {
  __builtin_amdgcn_global_load_async_to_lds_b128((gptr128)(g), (lptr128)(l), 0, 0);
}
__device__ __forceinline__ void async_wait0() {
#if __has_builtin(__builtin_amdgcn_s_wait_asynccnt)
  __builtin_amdgcn_s_wait_asynccnt(0);
#else
  asm volatile("s_wait_asynccnt 0x0" ::: "memory");
#endif
}
#endif

// ---------------------------------------------------------------------------
// ds_swizzle xor-butterfly within 16-lane groups (offset = xor<<10 | and 0x1f)
// ---------------------------------------------------------------------------
template <int OFF>
__device__ __forceinline__ float swz(float x) {
  return __int_as_float(__builtin_amdgcn_ds_swizzle(__float_as_int(x), OFF));
}
__device__ __forceinline__ void redmax8(float a[8]) {
#pragma unroll
  for (int r = 0; r < 8; r++) a[r] = fmaxf(a[r], swz<0x041F>(a[r]));
#pragma unroll
  for (int r = 0; r < 8; r++) a[r] = fmaxf(a[r], swz<0x081F>(a[r]));
#pragma unroll
  for (int r = 0; r < 8; r++) a[r] = fmaxf(a[r], swz<0x101F>(a[r]));
#pragma unroll
  for (int r = 0; r < 8; r++) a[r] = fmaxf(a[r], swz<0x201F>(a[r]));
}
__device__ __forceinline__ void redsum8(float a[8]) {
#pragma unroll
  for (int r = 0; r < 8; r++) a[r] += swz<0x041F>(a[r]);
#pragma unroll
  for (int r = 0; r < 8; r++) a[r] += swz<0x081F>(a[r]);
#pragma unroll
  for (int r = 0; r < 8; r++) a[r] += swz<0x101F>(a[r]);
#pragma unroll
  for (int r = 0; r < 8; r++) a[r] += swz<0x201F>(a[r]);
}

// ---------------------------------------------------------------------------
// fp32 -> bf16 conversion (weights), vectorized: 8 elems/thread
// (n is always a multiple of 8; buffers are 16B aligned)
// ---------------------------------------------------------------------------
__global__ __launch_bounds__(256) void cvt_f32_bf16(
    const float* __restrict__ in, u16* __restrict__ out, int n8) {
  int i = blockIdx.x * blockDim.x + threadIdx.x;  // chunk of 8
  if (i >= n8) return;
  const float4* src = (const float4*)(in) + 2 * (size_t)i;
  float4 f0 = src[0];
  float4 f1 = src[1];
  U16x8 r;
  r.u[0] = f_to_bf(f0.x); r.u[1] = f_to_bf(f0.y);
  r.u[2] = f_to_bf(f0.z); r.u[3] = f_to_bf(f0.w);
  r.u[4] = f_to_bf(f1.x); r.u[5] = f_to_bf(f1.y);
  r.u[6] = f_to_bf(f1.z); r.u[7] = f_to_bf(f1.w);
  ((uint4*)out)[i] = r.q;
}

// ---------------------------------------------------------------------------
// embedding gather + concat -> bf16 [B*S, 2E], vectorized 4 elems/thread/side
// ---------------------------------------------------------------------------
__global__ __launch_bounds__(256) void gather_concat(
    const int* __restrict__ item, const int* __restrict__ skill,
    const float* __restrict__ embI, const float* __restrict__ embS,
    u16* __restrict__ xcat) {
  int row = blockIdx.x;
  int it = item[row], sk = skill[row];
  const float4* pi = (const float4*)(embI + (size_t)it * 1024);
  const float4* ps = (const float4*)(embS + (size_t)sk * 1024);
  u16* out = xcat + (size_t)row * 2048;
  int c = threadIdx.x;  // 256 threads x 4 elems = 1024 per side
  {
    float4 f = pi[c];
    U16x4 r;
    r.u[0] = f_to_bf(f.x); r.u[1] = f_to_bf(f.y);
    r.u[2] = f_to_bf(f.z); r.u[3] = f_to_bf(f.w);
    ((uint2*)out)[c] = r.q;
  }
  {
    float4 f = ps[c];
    U16x4 r;
    r.u[0] = f_to_bf(f.x); r.u[1] = f_to_bf(f.y);
    r.u[2] = f_to_bf(f.z); r.u[3] = f_to_bf(f.w);
    ((uint2*)(out + 1024))[c] = r.q;
  }
}

// ---------------------------------------------------------------------------
// GEMM: C[M,N] = act( A[M,K](bf16) @ W[N,K]^T(bf16) + bias[N](f32) ), bf16 out
// 256 threads (8 waves), tile 128x128x32, double-buffered LDS; async
// global->LDS copies when available. ACT: 0 = none, 1 = exact GELU.
// ---------------------------------------------------------------------------
template <int ACT>
__global__ __launch_bounds__(256) void gemm_bf16(
    const u16* __restrict__ A, const u16* __restrict__ W,
    const float* __restrict__ bias, u16* __restrict__ Cout,
    int M, int N, int K) {
  constexpr int BM = 128, BN = 128, BK = 32, LDP = 40;  // 80B row stride
  __shared__ __align__(16) u16 sA[2][BM * LDP];
  __shared__ __align__(16) u16 sB[2][BN * LDP];

  const int tid  = threadIdx.x;
  const int lane = tid & 31;
  const int wave = tid >> 5;
  const int hi   = lane >> 4;
  const int lh   = lane & 15;
  const int wm0  = (wave & 3) * 32;   // 4 row-waves
  const int wn0  = (wave >> 2) * 64;  // 2 col-waves
  const int m0   = blockIdx.x * BM;
  const int n0   = blockIdx.y * BN;
  const int lrow = tid >> 1;
  const int lcol = (tid & 1) * 16;

  const u16* gA = A + (size_t)(m0 + lrow) * K + lcol;
  const u16* gW = W + (size_t)(n0 + lrow) * K + lcol;
  const int lofs = lrow * LDP + lcol;

  v8f acc[2][4] = {};

#if HAVE_ASYNC_LDS
  auto fetch = [&](int k0, int buf) {
    async_cp_b128(gA + k0,     &sA[buf][lofs]);
    async_cp_b128(gA + k0 + 8, &sA[buf][lofs + 8]);
    async_cp_b128(gW + k0,     &sB[buf][lofs]);
    async_cp_b128(gW + k0 + 8, &sB[buf][lofs + 8]);
  };
  fetch(0, 0);
  async_wait0();
  __syncthreads();
#else
  {
    uint4 a0 = *(const uint4*)(gA);
    uint4 a1 = *(const uint4*)(gA + 8);
    uint4 w0 = *(const uint4*)(gW);
    uint4 w1 = *(const uint4*)(gW + 8);
    *(uint4*)&sA[0][lofs]     = a0;
    *(uint4*)&sA[0][lofs + 8] = a1;
    *(uint4*)&sB[0][lofs]     = w0;
    *(uint4*)&sB[0][lofs + 8] = w1;
  }
  __syncthreads();
#endif

  int cur = 0;
  for (int k0 = 0; k0 < K; k0 += BK) {
    const int nxt = cur ^ 1;
    const bool hasNext = (k0 + BK) < K;
#if HAVE_ASYNC_LDS
    if (hasNext) fetch(k0 + BK, nxt);
#else
    uint4 a0, a1, w0, w1;
    if (hasNext) {
      const u16* pa = gA + k0 + BK;
      const u16* pw = gW + k0 + BK;
      __builtin_prefetch(pa + BK, 0, 1);
      __builtin_prefetch(pw + BK, 0, 1);
      a0 = *(const uint4*)(pa);
      a1 = *(const uint4*)(pa + 8);
      w0 = *(const uint4*)(pw);
      w1 = *(const uint4*)(pw + 8);
    }
#endif

    FragBF fa[2], fb[4];
#pragma unroll
    for (int i = 0; i < 2; i++) {  // A frag: lane<16 K 0-7/16-23, lane>=16 K 8-15/24-31
      const u16* p = &sA[cur][(wm0 + i * 16 + lh) * LDP + hi * 8];
      fa[i].q[0] = *(const uint4*)(p);
      fa[i].q[1] = *(const uint4*)(p + 16);
    }
#pragma unroll
    for (int j = 0; j < 4; j++) {  // B frag: col=lh, lane<16 K 0-7/16-23, lane>=16 K 8-15/24-31
      const u16* p = &sB[cur][(wn0 + j * 16 + lh) * LDP + hi * 8];
      fb[j].q[0] = *(const uint4*)(p);
      fb[j].q[1] = *(const uint4*)(p + 16);
    }
#pragma unroll
    for (int i = 0; i < 2; i++)
#pragma unroll
      for (int j = 0; j < 4; j++)
        acc[i][j] = WMMA_BF16(fa[i].v, fb[j].v, acc[i][j]);

#if HAVE_ASYNC_LDS
    async_wait0();
#else
    if (hasNext) {
      *(uint4*)&sA[nxt][lofs]     = a0;
      *(uint4*)&sA[nxt][lofs + 8] = a1;
      *(uint4*)&sB[nxt][lofs]     = w0;
      *(uint4*)&sB[nxt][lofs + 8] = w1;
    }
#endif
    __syncthreads();
    cur = nxt;
  }

  // epilogue: bias + activation + bf16 store
#pragma unroll
  for (int i = 0; i < 2; i++) {
#pragma unroll
    for (int j = 0; j < 4; j++) {
      int col = n0 + wn0 + j * 16 + lh;
      float bv = bias[col];
#pragma unroll
      for (int r = 0; r < 8; r++) {
        int row = m0 + wm0 + i * 16 + r + hi * 8;
        float x = acc[i][j][r] + bv;
        if (ACT == 1) x = 0.5f * x * (1.0f + erff(x * 0.70710678118654752f));
        Cout[(size_t)row * N + col] = f_to_bf(x);
      }
    }
  }
}

// ---------------------------------------------------------------------------
// Flash attention: O = softmax( Q (K+PE)^T / sqrt(D) ) V.
// glo dropped: per-query additive constant is softmax-invariant.
// 4 waves/block, each wave owns 16 query rows; 32-key chunks staged in LDS.
// ---------------------------------------------------------------------------
__global__ __launch_bounds__(128) void attn_flash(
    const u16* __restrict__ Q, const u16* __restrict__ Kb,
    const u16* __restrict__ Vb, const float* __restrict__ pe,
    u16* __restrict__ O) {
  constexpr int S = 1024, E = 1024, D = 64, KT = 32;
  constexpr int LK = 72;  // 144B row stride
  constexpr int LV = 40;  // 80B
  constexpr int LP = 40;
  __shared__ __align__(16) u16 sK[32 * LK];     // K' chunk [32 keys][64 d]
  __shared__ __align__(16) u16 sVt[64 * LV];    // V^T chunk [64 d][32 keys]
  __shared__ __align__(16) u16 sP[4][16 * LP];  // per-wave P [16 q][32 keys]

  const int tid  = threadIdx.x;
  const int lane = tid & 31;
  const int wave = tid >> 5;
  const int hi   = lane >> 4;
  const int lh   = lane & 15;
  const int h    = blockIdx.y;
  const int b    = blockIdx.z;
  const int qbase = blockIdx.x * 64 + wave * 16;

  FragBF qa[2];
  {
    size_t rowQ = ((size_t)(b * S + qbase + lh)) * E + h * D;
#pragma unroll
    for (int j = 0; j < 2; j++) {
      const u16* p = Q + rowQ + j * 32 + hi * 8;
      qa[j].q[0] = *(const uint4*)(p);
      qa[j].q[1] = *(const uint4*)(p + 16);
    }
  }

  float mrow[8], lsum[8];
  v8f o[4] = {};
#pragma unroll
  for (int r = 0; r < 8; r++) { mrow[r] = -1e30f; lsum[r] = 0.0f; }
  const float scale = 0.125f;  // 1/sqrt(64)

  // cooperative loader mapping: 128 threads; 16 elems each of K' and V
  const int cr = tid >> 2;         // key row in chunk 0..31
  const int cc = (tid & 3) * 16;   // d col base

  for (int kt = 0; kt < S; kt += KT) {
    {
      size_t g = ((size_t)(b * S + kt + cr)) * E + h * D + cc;
      const float* pep = pe + (size_t)(kt + cr) * D + cc;
      U16x8 k0u, k1u, v0u, v1u, e0, e1;
      k0u.q = *(const uint4*)(Kb + g);
      k1u.q = *(const uint4*)(Kb + g + 8);
      v0u.q = *(const uint4*)(Vb + g);
      v1u.q = *(const uint4*)(Vb + g + 8);
#pragma unroll
      for (int c = 0; c < 8; c++) {
        e0.u[c] = f_to_bf(bf_to_f(k0u.u[c]) + pep[c]);
        e1.u[c] = f_to_bf(bf_to_f(k1u.u[c]) + pep[8 + c]);
      }
      *(uint4*)&sK[cr * LK + cc]     = e0.q;
      *(uint4*)&sK[cr * LK + cc + 8] = e1.q;
#pragma unroll
      for (int c = 0; c < 8; c++) {
        sVt[(cc + c) * LV + cr]     = v0u.u[c];
        sVt[(cc + 8 + c) * LV + cr] = v1u.u[c];
      }
    }
    __syncthreads();

    // scores: 2 tiles of 16 keys, chained over 2 d-chunks (WMMA K=32)
    v8f s0, s1;
    {
      FragBF b0, b1;
#pragma unroll
      for (int t = 0; t < 2; t++) {
        const u16* p0 = &sK[(t * 16 + lh) * LK + hi * 8];
        b0.q[0] = *(const uint4*)(p0);
        b0.q[1] = *(const uint4*)(p0 + 16);
        const u16* p1 = &sK[(t * 16 + lh) * LK + 32 + hi * 8];
        b1.q[0] = *(const uint4*)(p1);
        b1.q[1] = *(const uint4*)(p1 + 16);
        v8f cc2 = {};
        cc2 = WMMA_BF16(qa[0].v, b0.v, cc2);
        cc2 = WMMA_BF16(qa[1].v, b1.v, cc2);
        if (t == 0) s0 = cc2; else s1 = cc2;
      }
    }

    // online softmax (rows striped: reg r -> row r (lanes<16) / r+8 (lanes>=16))
    float mx[8], rs[8], corr[8];
#pragma unroll
    for (int r = 0; r < 8; r++) {
      float v0 = s0[r] * scale, v1 = s1[r] * scale;
      s0[r] = v0; s1[r] = v1;
      mx[r] = fmaxf(v0, v1);
    }
    redmax8(mx);
#pragma unroll
    for (int r = 0; r < 8; r++) {
      float nm = fmaxf(mrow[r], mx[r]);
      corr[r] = __expf(mrow[r] - nm);
      mrow[r] = nm;
      float p0 = __expf(s0[r] - nm);
      float p1 = __expf(s1[r] - nm);
      s0[r] = p0; s1[r] = p1;
      rs[r] = p0 + p1;
    }
    redsum8(rs);
#pragma unroll
    for (int r = 0; r < 8; r++) lsum[r] = lsum[r] * corr[r] + rs[r];
#pragma unroll
    for (int t = 0; t < 4; t++)
#pragma unroll
      for (int r = 0; r < 8; r++) o[t][r] *= corr[r];

    // P -> LDS for A-fragment re-layout
#pragma unroll
    for (int r = 0; r < 8; r++) {
      int row = r + hi * 8;
      sP[wave][row * LP + lh]      = f_to_bf(s0[r]);
      sP[wave][row * LP + 16 + lh] = f_to_bf(s1[r]);
    }
    FragBF pa;
    {
      const u16* p = &sP[wave][lh * LP + hi * 8];
      pa.q[0] = *(const uint4*)(p);
      pa.q[1] = *(const uint4*)(p + 16);
    }
#pragma unroll
    for (int t = 0; t < 4; t++) {
      FragBF bvf;
      const u16* p = &sVt[(t * 16 + lh) * LV + hi * 8];
      bvf.q[0] = *(const uint4*)(p);
      bvf.q[1] = *(const uint4*)(p + 16);
      o[t] = WMMA_BF16(pa.v, bvf.v, o[t]);
    }
    __syncthreads();
  }

  // normalize + store bf16
#pragma unroll
  for (int t = 0; t < 4; t++) {
#pragma unroll
    for (int r = 0; r < 8; r++) {
      int row = qbase + r + hi * 8;
      int col = h * D + t * 16 + lh;
      O[((size_t)(b * S + row)) * E + col] = f_to_bf(o[t][r] / lsum[r]);
    }
  }
}

// ---------------------------------------------------------------------------
// final projection: out[row] = sum_e x[row,e]*W_out[e] + b_out
// ---------------------------------------------------------------------------
__global__ __launch_bounds__(256) void out_proj(
    const u16* __restrict__ x, const float* __restrict__ w,
    const float* __restrict__ b, float* __restrict__ out) {
  __shared__ float red[256];
  int row = blockIdx.x;
  float s = 0.0f;
  for (int c = threadIdx.x; c < 1024; c += 256)
    s += bf_to_f(x[(size_t)row * 1024 + c]) * w[c];
  red[threadIdx.x] = s;
  __syncthreads();
  for (int k = 128; k > 0; k >>= 1) {
    if (threadIdx.x < k) red[threadIdx.x] += red[threadIdx.x + k];
    __syncthreads();
  }
  if (threadIdx.x == 0) out[row] = red[0] + b[0];
}

// ---------------------------------------------------------------------------
extern "C" void kernel_launch(void* const* d_in, const int* in_sizes, int n_in,
                              void* d_out, int out_size, void* d_ws, size_t ws_size,
                              hipStream_t stream) {
  constexpr int B = 4, S = 1024, E = 1024, L = 4, MAX_POS = 1024, D = 64;
  constexpr int M = B * S;  // 4096

  const int*   item      = (const int*)d_in[0];
  const int*   skill     = (const int*)d_in[1];
  const float* emb_item  = (const float*)d_in[5];
  const float* emb_skill = (const float*)d_in[6];
  const float* W_in      = (const float*)d_in[7];
  const float* b_in      = (const float*)d_in[8];
  const float* Wq        = (const float*)d_in[9];
  const float* bq        = (const float*)d_in[10];
  const float* Wk        = (const float*)d_in[11];
  const float* bk        = (const float*)d_in[12];
  const float* Wv        = (const float*)d_in[13];
  const float* bv        = (const float*)d_in[14];
  const float* pos_key   = (const float*)d_in[17];
  const float* Wl        = (const float*)d_in[18];
  const float* bl        = (const float*)d_in[19];
  const float* W_out     = (const float*)d_in[20];
  const float* b_out     = (const float*)d_in[21];

  char* ws = (char*)d_ws;
  size_t off = 0;
  auto alloc_u16 = [&](size_t elems) {
    u16* p = (u16*)(ws + off);
    off += elems * sizeof(u16);
    return p;
  };
  u16* wbf_in = alloc_u16((size_t)E * 2 * E);
  u16* wbf_q  = alloc_u16((size_t)L * E * E);
  u16* wbf_k  = alloc_u16((size_t)L * E * E);
  u16* wbf_v  = alloc_u16((size_t)L * E * E);
  u16* wbf_l  = alloc_u16((size_t)L * 3 * E * E);
  u16* xcat   = alloc_u16((size_t)M * 2 * E);
  u16* xb     = alloc_u16((size_t)M * E);
  u16* q0b    = alloc_u16((size_t)M * E);
  u16* kb     = alloc_u16((size_t)M * E);
  u16* vb     = alloc_u16((size_t)M * E);
  u16* o0     = alloc_u16((size_t)M * E);
  u16* o1     = alloc_u16((size_t)M * E);
  (void)ws_size; (void)n_in; (void)in_sizes; (void)out_size;

  auto cvt = [&](const float* src, u16* dst, int n) {
    int n8 = n / 8;
    cvt_f32_bf16<<<(n8 + 255) / 256, 256, 0, stream>>>(src, dst, n8);
  };
  cvt(W_in, wbf_in, E * 2 * E);
  cvt(Wq, wbf_q, L * E * E);
  cvt(Wk, wbf_k, L * E * E);
  cvt(Wv, wbf_v, L * E * E);
  cvt(Wl, wbf_l, L * 3 * E * E);

  gather_concat<<<M, 256, 0, stream>>>(item, skill, emb_item, emb_skill, xcat);

  dim3 gemm_grid(M / 128, E / 128);
  gemm_bf16<0><<<gemm_grid, 256, 0, stream>>>(xcat, wbf_in, b_in, xb, M, E, 2 * E);

  for (int l = 0; l < L; l++) {
    const size_t wo = (size_t)l * E * E;
    gemm_bf16<0><<<gemm_grid, 256, 0, stream>>>(xb,  wbf_q + wo, bq + l * E, q0b, M, E, E);
    gemm_bf16<0><<<gemm_grid, 256, 0, stream>>>(q0b, wbf_k + wo, bk + l * E, kb,  M, E, E);
    gemm_bf16<0><<<gemm_grid, 256, 0, stream>>>(q0b, wbf_v + wo, bv + l * E, vb,  M, E, E);

    attn_flash<<<dim3(S / 64, E / D, B), 128, 0, stream>>>(
        q0b, kb, vb, pos_key + (size_t)l * MAX_POS * D, o0);

    const size_t lo = (size_t)l * 3 * E * E;
    gemm_bf16<1><<<gemm_grid, 256, 0, stream>>>(o0, wbf_l + lo,                     bl + (l * 3 + 0) * E, o1, M, E, E);
    gemm_bf16<1><<<gemm_grid, 256, 0, stream>>>(o1, wbf_l + lo + (size_t)E * E,     bl + (l * 3 + 1) * E, o0, M, E, E);
    gemm_bf16<1><<<gemm_grid, 256, 0, stream>>>(o0, wbf_l + lo + (size_t)2 * E * E, bl + (l * 3 + 2) * E, xb, M, E, E);
  }

  out_proj<<<M, 256, 0, stream>>>(xb, W_out, b_out, (float*)d_out);
}
